// DifferentialAttention_81535659147992
// MI455X (gfx1250) — matrix operand order repaired and checked by reading, not compiled, not run on hardware
//
#include <hip/hip_runtime.h>
#include <hip/hip_bf16.h>

#define SEQ 2048
#define DIM 2048
#define QKD 4096
#define L_INIT 0.35550906759096927f   // 0.8 - 0.6*exp(-0.3)
#define EPS_RMS 1e-6f

typedef __attribute__((ext_vector_type(4)))  float  f32x4;
typedef __attribute__((ext_vector_type(8)))  float  f32x8;
typedef __attribute__((ext_vector_type(8)))  __bf16 bf16x8;
typedef __attribute__((ext_vector_type(16))) __bf16 bf16x16;

static __device__ __forceinline__ f32x8 wmma_bf16(bf16x16 a, bf16x16 b, f32x8 c) {
  return __builtin_amdgcn_wmma_f32_16x16x32_bf16(false, a, false, b, (short)0, c, false, false);
}

// two contiguous 16-byte bf16 chunks -> one 16-elem fragment
static __device__ __forceinline__ bf16x16 ld2x8(const __bf16* p0, const __bf16* p1) {
  bf16x8 lo = *(const bf16x8*)p0;
  bf16x8 hi = *(const bf16x8*)p1;
  return __builtin_shufflevector(lo, hi, 0,1,2,3,4,5,6,7,8,9,10,11,12,13,14,15);
}
// A-frag (ISA §7.12.2): per-lane chunks at +0 and +16 elems
static __device__ __forceinline__ bf16x16 load_fragA(const __bf16* p) { return ld2x8(p, p + 16); }
// B-frag: per-lane chunks at +0 and +16 elems
static __device__ __forceinline__ bf16x16 load_fragB(const __bf16* p) { return ld2x8(p, p + 16); }

// -------------------- fp32 -> bf16 bulk convert ------------------------------
__global__ __launch_bounds__(256) void cvt_f32_bf16(const float* __restrict__ src,
                                                    __bf16* __restrict__ dst, int n) {
  int i = (blockIdx.x * 256 + threadIdx.x) * 8;
  if (i < n) {
    f32x4 a = *(const f32x4*)(src + i);
    f32x4 b = *(const f32x4*)(src + i + 4);
    bf16x8 r;
#pragma unroll
    for (int j = 0; j < 4; ++j) { r[j] = (__bf16)a[j]; r[j + 4] = (__bf16)b[j]; }
    *(bf16x8*)(dst + i) = r;
  }
}

// -------------------- C = A[M,K] * W[N,K]^T  (bf16 WMMA, f32 acc) ------------
// block: 256 threads = 8 waves (2 M x 4 N), block tile 64 x 256, wave tile 32x64
template <typename CT>
__global__ __launch_bounds__(256) void gemm_xwT(const __bf16* __restrict__ A,
                                                const __bf16* __restrict__ W,
                                                CT* __restrict__ C,
                                                int M, int N, int K) {
  const int lane = threadIdx.x & 31;
  const int wave = threadIdx.x >> 5;
  const int half = lane >> 4;
  const int l16  = lane & 15;
  const int mBase = blockIdx.y * 64  + (wave >> 2) * 32;
  const int nBase = blockIdx.x * 256 + (wave & 3) * 64;

  f32x8 acc[2][4];
#pragma unroll
  for (int mt = 0; mt < 2; ++mt)
#pragma unroll
    for (int nt = 0; nt < 4; ++nt)
#pragma unroll
      for (int r = 0; r < 8; ++r) acc[mt][nt][r] = 0.f;

  const __bf16* pa0 = A + (size_t)(mBase + l16) * K + half * 8;
  const __bf16* pa1 = A + (size_t)(mBase + 16 + l16) * K + half * 8;
  const __bf16* pw0 = W + (size_t)(nBase + l16) * K + half * 8;

  for (int kk = 0; kk < K; kk += 32) {
    bf16x16 af[2], bfr[4];
    af[0] = load_fragA(pa0 + kk);
    af[1] = load_fragA(pa1 + kk);
#pragma unroll
    for (int nt = 0; nt < 4; ++nt)
      bfr[nt] = load_fragB(pw0 + (size_t)nt * 16 * K + kk);
#pragma unroll
    for (int mt = 0; mt < 2; ++mt)
#pragma unroll
      for (int nt = 0; nt < 4; ++nt)
        acc[mt][nt] = wmma_bf16(af[mt], bfr[nt], acc[mt][nt]);
  }
#pragma unroll
  for (int mt = 0; mt < 2; ++mt)
#pragma unroll
    for (int nt = 0; nt < 4; ++nt)
#pragma unroll
      for (int r = 0; r < 8; ++r) {
        int row = mBase + mt * 16 + r + half * 8;
        int col = nBase + nt * 16 + l16;
        C[(size_t)row * N + col] = (CT)acc[mt][nt][r];
      }
}

// -------------------- V -> V^T (bf16, LDS tiled) -----------------------------
__global__ void transpose_bf16(const __bf16* __restrict__ src, __bf16* __restrict__ dst,
                               int rows, int cols) {
  __shared__ __bf16 tile[32][33];
  const int c0 = blockIdx.x * 32, r0 = blockIdx.y * 32;
  const int tx = threadIdx.x, ty = threadIdx.y;
  for (int i = ty; i < 32; i += 8) tile[i][tx] = src[(size_t)(r0 + i) * cols + c0 + tx];
  __syncthreads();
  for (int i = ty; i < 32; i += 8) dst[(size_t)(c0 + i) * rows + r0 + tx] = tile[tx][i];
}

// -------------------- lambda scalar ------------------------------------------
__global__ void lambda_kernel(const float* a, const float* b, const float* c,
                              const float* d, float* out) {
  if (threadIdx.x == 0) {
    float s1 = 0.f, s2 = 0.f;
#pragma unroll
    for (int i = 0; i < 8; ++i) { s1 += a[i] * b[i]; s2 += c[i] * d[i]; }
    *out = __expf(s1) - __expf(s2) + L_INIT;
  }
}

// ---- one softmax component for a 32-key tile: logits -> running softmax -----
// returns the P tile as a WMMA A-fragment (via per-wave LDS relayout)
static __device__ __forceinline__ bf16x16 attn_probs(
    const __bf16* __restrict__ Q, const __bf16* __restrict__ Kc,
    __bf16* myscr, int qBase, int qc, int kb, int l16, int half,
    f32x8 (&acc)[16], float (&mst)[8], float (&lst)[8]) {
  f32x8 s0, s1;
#pragma unroll
  for (int r = 0; r < 8; ++r) { s0[r] = 0.f; s1[r] = 0.f; }
  // logits: Q(16x256) @ K_tile^T(256x32), dh chunks of 32
#pragma unroll
  for (int d = 0; d < 8; ++d) {
    const __bf16* qp = Q + (size_t)(qBase + l16) * QKD + qc + d * 32 + half * 8;
    bf16x16 af = load_fragA(qp);
    const __bf16* k0 = Kc + (size_t)(kb + l16) * QKD + qc + d * 32 + half * 8;
    const __bf16* k1 = Kc + (size_t)(kb + 16 + l16) * QKD + qc + d * 32 + half * 8;
    s0 = wmma_bf16(af, load_fragB(k0), s0);
    s1 = wmma_bf16(af, load_fragB(k1), s1);
  }
  float rmax[8], rsum[8];
#pragma unroll
  for (int r = 0; r < 8; ++r) {  // scale 1/sqrt(256) + causal mask
    int m = qBase + r + half * 8;
    float v0 = (kb + l16      <= m) ? s0[r] * 0.0625f : -1e30f;
    float v1 = (kb + 16 + l16 <= m) ? s1[r] * 0.0625f : -1e30f;
    s0[r] = v0; s1[r] = v1;
    rmax[r] = fmaxf(v0, v1);
  }
#pragma unroll
  for (int off = 1; off < 16; off <<= 1)
#pragma unroll
    for (int r = 0; r < 8; ++r) rmax[r] = fmaxf(rmax[r], __shfl_xor(rmax[r], off, 32));
#pragma unroll
  for (int r = 0; r < 8; ++r) {
    float nm    = fmaxf(mst[r], rmax[r]);
    float alpha = __expf(mst[r] - nm);
    float p0 = __expf(s0[r] - nm);
    float p1 = __expf(s1[r] - nm);
    s0[r] = p0; s1[r] = p1;
    rsum[r] = p0 + p1;
    mst[r] = nm;
    lst[r] *= alpha;
#pragma unroll
    for (int nt = 0; nt < 16; ++nt) acc[nt][r] *= alpha;
  }
#pragma unroll
  for (int off = 1; off < 16; off <<= 1)
#pragma unroll
    for (int r = 0; r < 8; ++r) rsum[r] += __shfl_xor(rsum[r], off, 32);
#pragma unroll
  for (int r = 0; r < 8; ++r) lst[r] += rsum[r];
  // relayout P: C-layout (regs) -> row-major LDS -> A-fragment
#pragma unroll
  for (int r = 0; r < 8; ++r) {
    myscr[(r + half * 8) * 32 + l16]      = (__bf16)s0[r];
    myscr[(r + half * 8) * 32 + 16 + l16] = (__bf16)s1[r];
  }
  return load_fragA(myscr + l16 * 32 + half * 8);
}

// -------------------- differential attention + GroupRMSNorm ------------------
// grid (SEQ/64, 8 heads), block 128 = 4 waves, wave tile = 16 query rows
__global__ __launch_bounds__(128) void diff_attn(const __bf16* __restrict__ Q,
                                                 const __bf16* __restrict__ Kc,
                                                 const __bf16* __restrict__ Vt,
                                                 const float* __restrict__ lam_p,
                                                 __bf16* __restrict__ O) {
  __shared__ __bf16 scr[4][2][16 * 32];
  const int lane = threadIdx.x & 31;
  const int wave = threadIdx.x >> 5;
  const int half = lane >> 4;
  const int l16  = lane & 15;
  const int head = blockIdx.y;
  const int qBase = blockIdx.x * 64 + wave * 16;
  const int col1 = (head >> 1) * 512 + (head & 1) * 256;  // component-1 q/k cols
  const int col2 = col1 + 2048;                           // component-2 q/k cols
  const int vcol = head * 256;
  const float lam = *lam_p;

  f32x8 acc1[16], acc2[16];
  float m1[8], l1[8], m2[8], l2[8];
#pragma unroll
  for (int nt = 0; nt < 16; ++nt)
#pragma unroll
    for (int r = 0; r < 8; ++r) { acc1[nt][r] = 0.f; acc2[nt][r] = 0.f; }
#pragma unroll
  for (int r = 0; r < 8; ++r) { m1[r] = -1e30f; m2[r] = -1e30f; l1[r] = 0.f; l2[r] = 0.f; }

  const int numKT = (qBase + 15) / 32 + 1;  // causal bound for this wave tile
  for (int kt = 0; kt < numKT; ++kt) {
    const int kb = kt * 32;
    if (kt + 1 < numKT) {  // prefetch next key tile (K rows of both components + V^T)
      const __bf16* nk = Kc + (size_t)(kb + 32 + l16) * QKD;
      __builtin_prefetch(nk + col1, 0, 0);
      __builtin_prefetch(nk + col2, 0, 0);
      __builtin_prefetch(Vt + (size_t)(vcol + l16) * SEQ + kb + 32, 0, 0);
    }
    bf16x16 pf1 = attn_probs(Q, Kc, scr[wave][0], qBase, col1, kb, l16, half, acc1, m1, l1);
    bf16x16 pf2 = attn_probs(Q, Kc, scr[wave][1], qBase, col2, kb, l16, half, acc2, m2, l2);
    // shared V tile: each B-frag feeds both components' accumulators
#pragma unroll
    for (int nt = 0; nt < 16; ++nt) {
      const __bf16* vp = Vt + (size_t)(vcol + nt * 16 + l16) * SEQ + kb + half * 8;
      bf16x16 vf = load_fragB(vp);
      acc1[nt] = wmma_bf16(pf1, vf, acc1[nt]);
      acc2[nt] = wmma_bf16(pf2, vf, acc2[nt]);
    }
  }

  // combine: softmax1 - lambda * softmax2
#pragma unroll
  for (int nt = 0; nt < 16; ++nt)
#pragma unroll
    for (int r = 0; r < 8; ++r)
      acc1[nt][r] = acc1[nt][r] / l1[r] - lam * (acc2[nt][r] / l2[r]);

  // GroupRMSNorm over groups of 32 cols (pairs of N-tiles), then *(1-L_INIT)
  const float oscale = 1.0f - L_INIT;
#pragma unroll
  for (int g = 0; g < 8; ++g) {
    float ss[8];
#pragma unroll
    for (int r = 0; r < 8; ++r)
      ss[r] = acc1[2 * g][r] * acc1[2 * g][r] + acc1[2 * g + 1][r] * acc1[2 * g + 1][r];
#pragma unroll
    for (int off = 1; off < 16; off <<= 1)
#pragma unroll
      for (int r = 0; r < 8; ++r) ss[r] += __shfl_xor(ss[r], off, 32);
#pragma unroll
    for (int r = 0; r < 8; ++r) {
      float rr = rsqrtf(ss[r] * (1.0f / 32.0f) + EPS_RMS) * oscale;
      acc1[2 * g][r] *= rr;
      acc1[2 * g + 1][r] *= rr;
    }
  }
#pragma unroll
  for (int nt = 0; nt < 16; ++nt)
#pragma unroll
    for (int r = 0; r < 8; ++r) {
      int m = qBase + r + half * 8;
      O[(size_t)m * DIM + vcol + nt * 16 + l16] = (__bf16)acc1[nt][r];
    }
}

// -------------------- host-side orchestration --------------------------------
extern "C" void kernel_launch(void* const* d_in, const int* in_sizes, int n_in,
                              void* d_out, int out_size, void* d_ws, size_t ws_size,
                              hipStream_t stream) {
  (void)in_sizes; (void)n_in; (void)out_size; (void)ws_size;
  const float* x   = (const float*)d_in[0];
  const float* lq1 = (const float*)d_in[1];
  const float* lk1 = (const float*)d_in[2];
  const float* lq2 = (const float*)d_in[3];
  const float* lk2 = (const float*)d_in[4];
  const float* wq  = (const float*)d_in[5];
  const float* wk  = (const float*)d_in[6];
  const float* wv  = (const float*)d_in[7];
  const float* wo  = (const float*)d_in[8];
  float* out = (float*)d_out;

  char* ws = (char*)d_ws;
  float*  lam = (float*)ws;                        // 4 B (+ pad to 256)
  __bf16* xb  = (__bf16*)(ws + 256);               // [SEQ, DIM]      8 MiB
  __bf16* wqb = xb  + (size_t)SEQ * DIM;           // [QKD, DIM]     16 MiB
  __bf16* wkb = wqb + (size_t)QKD * DIM;           // [QKD, DIM]     16 MiB
  __bf16* wvb = wkb + (size_t)QKD * DIM;           // [DIM, DIM]      8 MiB
  __bf16* wob = wvb + (size_t)DIM * DIM;           // [DIM, DIM]      8 MiB
  __bf16* qb  = wob + (size_t)DIM * DIM;           // [SEQ, QKD]     16 MiB
  __bf16* kbf = qb  + (size_t)SEQ * QKD;           // [SEQ, QKD]     16 MiB
  __bf16* vb  = kbf + (size_t)SEQ * QKD;           // [SEQ, DIM]      8 MiB
  __bf16* vt  = vb  + (size_t)SEQ * DIM;           // [DIM, SEQ]      8 MiB
  __bf16* ob  = vt  + (size_t)SEQ * DIM;           // [SEQ, DIM]      8 MiB

  lambda_kernel<<<1, 32, 0, stream>>>(lq1, lk1, lq2, lk2, lam);

  const int nx = SEQ * DIM, nqk = QKD * DIM;
  cvt_f32_bf16<<<nx  / 2048, 256, 0, stream>>>(x,  xb,  nx);
  cvt_f32_bf16<<<nqk / 2048, 256, 0, stream>>>(wq, wqb, nqk);
  cvt_f32_bf16<<<nqk / 2048, 256, 0, stream>>>(wk, wkb, nqk);
  cvt_f32_bf16<<<nx  / 2048, 256, 0, stream>>>(wv, wvb, nx);
  cvt_f32_bf16<<<nx  / 2048, 256, 0, stream>>>(wo, wob, nx);

  gemm_xwT<__bf16><<<dim3(QKD / 256, SEQ / 64), 256, 0, stream>>>(xb, wqb, qb,  SEQ, QKD, DIM);
  gemm_xwT<__bf16><<<dim3(QKD / 256, SEQ / 64), 256, 0, stream>>>(xb, wkb, kbf, SEQ, QKD, DIM);
  gemm_xwT<__bf16><<<dim3(DIM / 256, SEQ / 64), 256, 0, stream>>>(xb, wvb, vb,  SEQ, DIM, DIM);
  transpose_bf16<<<dim3(DIM / 32, SEQ / 32), dim3(32, 8), 0, stream>>>(vb, vt, SEQ, DIM);
  diff_attn<<<dim3(SEQ / 64, 8), 128, 0, stream>>>(qb, kbf, vt, lam, ob);
  gemm_xwT<float><<<dim3(DIM / 256, SEQ / 64), 256, 0, stream>>>(ob, wob, out, SEQ, DIM, DIM);
}
